// FastASTEncoder_4458176053344
// MI455X (gfx1250) — matrix operand order repaired and checked by reading, not compiled, not run on hardware
//
#include <hip/hip_runtime.h>

// ---------------------------------------------------------------------------
// FastASTEncoder for MI455X (gfx1250, wave32).
//  * GEMMs: v_wmma_f32_16x16x32_bf16, 128x128 block tile, 64x32 wave tile
//    (8 WMMAs per K-step per wave). Weights pre-transposed+converted to bf16
//    [N,K]; activations pre-converted to bf16, so tile staging is pure copies
//    routed through GLOBAL_LOAD_ASYNC_TO_LDS_B128 (ASYNCcnt) when available.
//  * Attention: one wave per (b,h,n); gathered k/v head rows are coalesced
//    256B loads; shfl-reduced disentangled scores; softmax over P=7.
//  * Residual+LayerNorm: one wave per 512-wide row.
// ---------------------------------------------------------------------------

#define B_    4
#define N_    1024
#define HID_  512
#define H_    8
#define DK_   64
#define P_    7
#define PARH_ 4
#define FFN_  2048
#define ROWS_ (B_ * N_)   // 4096

typedef __attribute__((ext_vector_type(16))) __bf16 v16bf;
typedef __attribute__((ext_vector_type(8)))  float  v8f;
typedef __attribute__((ext_vector_type(4)))  int    v4i;
typedef unsigned short u16;
typedef __attribute__((address_space(1))) v4i gv4i_t;   // global int4
typedef __attribute__((address_space(3))) v4i lv4i_t;   // LDS int4

__device__ inline u16 f2bfu(float f) {              // f32 -> bf16 bits, RNE
    union { float f; unsigned u; } x; x.f = f;
    return (u16)((x.u + 0x7FFFu + ((x.u >> 16) & 1u)) >> 16);
}

// 16-byte global -> LDS copy; async DMA path (ASYNCcnt) when the builtin is
// available, plain vector copy otherwise.
__device__ inline void cp16_g2l(const void* g, void* l) {
#if defined(__has_builtin) && __has_builtin(__builtin_amdgcn_global_load_async_to_lds_b128)
    __builtin_amdgcn_global_load_async_to_lds_b128((gv4i_t*)(void*)g, (lv4i_t*)l, 0, 0);
#else
    *(uint4*)l = *(const uint4*)g;
#endif
}

__device__ inline void wait_async_lds() {
    asm volatile("s_wait_asynccnt 0x0" ::: "memory");
}

// ---------------------------------------------------------------------------
// Elementwise f32 -> bf16 (8 elements / thread, vectorized)
// ---------------------------------------------------------------------------
__global__ __launch_bounds__(256)
void f2bf_kernel(const float* __restrict__ in, u16* __restrict__ out, int n8) {
    const int i = blockIdx.x * blockDim.x + threadIdx.x;
    if (i >= n8) return;
    const float4* p = (const float4*)in + (size_t)i * 2;
    float4 a = p[0], b = p[1];
    union { u16 h[8]; uint4 u; } pk;
    pk.h[0] = f2bfu(a.x); pk.h[1] = f2bfu(a.y); pk.h[2] = f2bfu(a.z); pk.h[3] = f2bfu(a.w);
    pk.h[4] = f2bfu(b.x); pk.h[5] = f2bfu(b.y); pk.h[6] = f2bfu(b.z); pk.h[7] = f2bfu(b.w);
    ((uint4*)out)[i] = pk.u;
}

// ---------------------------------------------------------------------------
// Weight prep: W[K,N] f32 (row-major)  ->  Wt[N,K] bf16 (row-major)
// 32x32 LDS-tiled transpose, coalesced on both sides.
// ---------------------------------------------------------------------------
__global__ __launch_bounds__(256)
void transpose_f2bf_kernel(const float* __restrict__ W, u16* __restrict__ Wt,
                           int K, int N) {
    __shared__ float tile[32][33];
    const int n0 = blockIdx.x * 32;
    const int k0 = blockIdx.y * 32;
    const int tx = threadIdx.x & 31;
    const int ty = threadIdx.x >> 5;            // 0..7
    #pragma unroll
    for (int r = ty; r < 32; r += 8)
        tile[r][tx] = W[(size_t)(k0 + r) * N + n0 + tx];
    __syncthreads();
    #pragma unroll
    for (int r = ty; r < 32; r += 8)
        Wt[(size_t)(n0 + r) * K + k0 + tx] = f2bfu(tile[tx][r]);
}

// ---------------------------------------------------------------------------
// WMMA GEMM:  C[M,N] = A[M,K](bf16) @ Wt[N,K](bf16)^T + bias  (optional ReLU)
// Block 128x128, BK=32. 8 waves: 2 (M) x 4 (N); wave tile 64x32 = 4x2 WMMA.
// ---------------------------------------------------------------------------
#define BM 128
#define BN 128
#define BK 32
#define LPAD 8   // 16B pad per LDS row

__global__ __launch_bounds__(256)
void gemm_bf16_wmma(const u16* __restrict__ A, const u16* __restrict__ Bt,
                    const float* __restrict__ bias, float* __restrict__ C,
                    int M, int K, int N, int relu) {
    __shared__ __bf16 Alds[BM][BK + LPAD];   // 128 x 40 bf16 (10KB)
    __shared__ __bf16 Blds[BN][BK + LPAD];   // 128 x 40 bf16, stored [n][k]

    const int tid    = threadIdx.x;
    const int lane   = tid & 31;
    const int waveId = tid >> 5;
    const int wm     = waveId >> 2;          // 0..1  (64-row M sub-tile)
    const int wn     = waveId & 3;           // 0..3  (32-col N sub-tile)
    const int row16  = lane & 15;
    const int khalf  = lane >> 4;

    const int m0 = blockIdx.y * BM;
    const int n0 = blockIdx.x * BN;

    v8f acc[4][2];
    const v8f vzero = {0.f, 0.f, 0.f, 0.f, 0.f, 0.f, 0.f, 0.f};
    #pragma unroll
    for (int mi = 0; mi < 4; ++mi)
        #pragma unroll
        for (int ni = 0; ni < 2; ++ni) acc[mi][ni] = vzero;

    // staging assignment: thread t -> row t/2, 16-elem half t%2 (32B)
    const int srow  = tid >> 1;
    const int shalf = tid & 1;
    const u16* aSrc = A  + (size_t)(m0 + srow) * K + shalf * 16;
    const u16* bSrc = Bt + (size_t)(n0 + srow) * K + shalf * 16;
    __bf16* aDst = &Alds[srow][shalf * 16];
    __bf16* bDst = &Blds[srow][shalf * 16];

    for (int k0 = 0; k0 < K; k0 += BK) {
        cp16_g2l(aSrc + k0,     aDst);
        cp16_g2l(aSrc + k0 + 8, aDst + 8);
        cp16_g2l(bSrc + k0,     bDst);
        cp16_g2l(bSrc + k0 + 8, bDst + 8);
        wait_async_lds();
        __syncthreads();

        // ---- fragments ----
        v16bf afrag[4], bfrag[2];
        #pragma unroll
        for (int mi = 0; mi < 4; ++mi) {
            const __bf16* ap = &Alds[wm * 64 + mi * 16 + row16][0];
            #pragma unroll
            for (int i = 0; i < 4; ++i) {
                int ks = khalf * 8 + 2 * i;          // ISA 16-bit A layout
                afrag[mi][2 * i]           = ap[ks];
                afrag[mi][2 * i + 1]       = ap[ks + 1];
                afrag[mi][2 * (i + 4)]     = ap[16 + ks];
                afrag[mi][2 * (i + 4) + 1] = ap[16 + ks + 1];
            }
        }
        #pragma unroll
        for (int ni = 0; ni < 2; ++ni) {
            const __bf16* bp = &Blds[wn * 32 + ni * 16 + row16][khalf * 8];
            #pragma unroll
            for (int i = 0; i < 16; ++i) bfrag[ni][i] = bp[i < 8 ? i : i + 8];
        }

        // ---- 8 WMMAs per K step ----
        #pragma unroll
        for (int mi = 0; mi < 4; ++mi)
            #pragma unroll
            for (int ni = 0; ni < 2; ++ni)
                acc[mi][ni] = __builtin_amdgcn_wmma_f32_16x16x32_bf16(
                    false, afrag[mi], false, bfrag[ni],
                    (short)0, acc[mi][ni], false, false);

        __syncthreads();
    }

    // ---- epilogue: bias (+ReLU), C/D VGPR layout scatter ----
    #pragma unroll
    for (int mi = 0; mi < 4; ++mi) {
        #pragma unroll
        for (int ni = 0; ni < 2; ++ni) {
            const int rbase = m0 + wm * 64 + mi * 16 + khalf * 8;
            const int col   = n0 + wn * 32 + ni * 16 + row16;
            const float bb  = bias ? bias[col] : 0.f;
            #pragma unroll
            for (int r = 0; r < 8; ++r) {
                float val = acc[mi][ni][r] + bb;
                if (relu) val = fmaxf(val, 0.f);
                C[(size_t)(rbase + r) * N + col] = val;
            }
        }
    }
}

// ---------------------------------------------------------------------------
// Relative attention: one wave per (b,h,n). Lanes stripe DK=64 (2 elems each).
// ---------------------------------------------------------------------------
__global__ __launch_bounds__(256)
void rel_attn_kernel(const float* __restrict__ q, const float* __restrict__ k,
                     const float* __restrict__ v,
                     const int* __restrict__ par, const int* __restrict__ bro,
                     const float* __restrict__ rel_q, const float* __restrict__ rel_k,
                     const float* __restrict__ rel_v, float* __restrict__ out) {
    const int wid = blockIdx.x * (blockDim.x >> 5) + (threadIdx.x >> 5);
    if (wid >= B_ * H_ * N_) return;
    const int lane = threadIdx.x & 31;
    const int n = wid % N_;
    const int h = (wid / N_) % H_;
    const int b = wid / (N_ * H_);

    const int d0 = lane, d1 = lane + 32;
    const float* qp = q + ((size_t)(b * N_ + n) * HID_) + h * DK_;
    const float qv0 = qp[d0], qv1 = qp[d1];
    const int* edges = (h < PARH_) ? par : bro;

    int   sidx[P_];
    float sc[P_];
    #pragma unroll
    for (int p = 0; p < P_; ++p) {
        const int s = edges[((size_t)b * P_ + p) * N_ + n];
        sidx[p] = s;
        const float* kp = k + ((size_t)(b * N_ + s) * HID_) + h * DK_;
        const float* rk = rel_k + ((size_t)h * P_ + p) * DK_;
        const float* rq = rel_q + ((size_t)h * P_ + p) * DK_;
        const float k0v = kp[d0], k1v = kp[d1];
        float part = qv0 * k0v + qv1 * k1v
                   + qv0 * rk[d0] + qv1 * rk[d1]
                   + rq[d0] * k0v + rq[d1] * k1v;
        #pragma unroll
        for (int off = 16; off > 0; off >>= 1) part += __shfl_xor(part, off, 32);
        sc[p] = part * 0.125f;                      // / sqrt(64)
    }

    float m = sc[0];
    #pragma unroll
    for (int p = 1; p < P_; ++p) m = fmaxf(m, sc[p]);
    float ssum = 0.f;
    #pragma unroll
    for (int p = 0; p < P_; ++p) { sc[p] = __expf(sc[p] - m); ssum += sc[p]; }
    const float inv = 1.f / ssum;

    float o0 = 0.f, o1 = 0.f;
    #pragma unroll
    for (int p = 0; p < P_; ++p) {
        const float a = sc[p] * inv;
        const float* vp = v + ((size_t)(b * N_ + sidx[p]) * HID_) + h * DK_;
        const float* rv = rel_v + ((size_t)h * P_ + p) * DK_;
        o0 += a * (vp[d0] + rv[d0]);
        o1 += a * (vp[d1] + rv[d1]);
    }
    float* op = out + ((size_t)(b * N_ + n) * HID_) + h * DK_;
    op[d0] = o0;
    op[d1] = o1;
}

// ---------------------------------------------------------------------------
// y = LayerNorm(x + delta) * s + b   — one wave per 512-wide row
// ---------------------------------------------------------------------------
__global__ __launch_bounds__(256)
void resid_ln_kernel(const float* __restrict__ x, const float* __restrict__ delta,
                     const float* __restrict__ s, const float* __restrict__ bprm,
                     float* __restrict__ y, int rows) {
    const int row = blockIdx.x * (blockDim.x >> 5) + (threadIdx.x >> 5);
    if (row >= rows) return;
    const int lane = threadIdx.x & 31;
    const float* xr = x + (size_t)row * HID_;
    const float* dr = delta ? delta + (size_t)row * HID_ : nullptr;

    float vals[16];
    float sum = 0.f, sq = 0.f;
    #pragma unroll
    for (int j = 0; j < 16; ++j) {
        const int c = lane + j * 32;
        float t = xr[c] + (dr ? dr[c] : 0.f);
        vals[j] = t;
        sum += t;
        sq  += t * t;
    }
    #pragma unroll
    for (int off = 16; off > 0; off >>= 1) {
        sum += __shfl_xor(sum, off, 32);
        sq  += __shfl_xor(sq,  off, 32);
    }
    const float mean = sum * (1.f / HID_);
    const float var  = sq * (1.f / HID_) - mean * mean;
    const float rinv = rsqrtf(var + 1e-5f);
    float* yr = y + (size_t)row * HID_;
    #pragma unroll
    for (int j = 0; j < 16; ++j) {
        const int c = lane + j * 32;
        yr[c] = (vals[j] - mean) * rinv * s[c] + bprm[c];
    }
}

// ---------------------------------------------------------------------------
// Host-side orchestration
// ---------------------------------------------------------------------------
extern "C" void kernel_launch(void* const* d_in, const int* in_sizes, int n_in,
                              void* d_out, int out_size, void* d_ws, size_t ws_size,
                              hipStream_t stream) {
    (void)in_sizes; (void)n_in; (void)out_size; (void)ws_size;

    const float* src_emb = (const float*)d_in[0];
    const int*   par     = (const int*)d_in[1];
    const int*   bro     = (const int*)d_in[2];
    const float* rel_q   = (const float*)d_in[3];
    const float* rel_k   = (const float*)d_in[4];
    const float* rel_v   = (const float*)d_in[5];
    const float* fln_s   = (const float*)d_in[6];
    const float* fln_b   = (const float*)d_in[7];

    char* wp = (char*)d_ws;
    auto alloc = [&](size_t bytes) -> void* {
        void* p = (void*)wp;
        wp += (bytes + 255) & ~(size_t)255;
        return p;
    };
    float* x    = (float*)alloc((size_t)ROWS_ * HID_ * 4);
    float* qb   = (float*)alloc((size_t)ROWS_ * HID_ * 4);
    float* kb   = (float*)alloc((size_t)ROWS_ * HID_ * 4);
    float* vb   = (float*)alloc((size_t)ROWS_ * HID_ * 4);
    float* attn = (float*)alloc((size_t)ROWS_ * HID_ * 4);
    float* tmp  = (float*)alloc((size_t)ROWS_ * HID_ * 4);
    float* h1   = (float*)alloc((size_t)ROWS_ * FFN_ * 4);
    u16*   xb   = (u16*)alloc((size_t)ROWS_ * FFN_ * 2);   // bf16 activations
    u16*   wt[2][6];
    for (int li = 0; li < 2; ++li) {
        wt[li][0] = (u16*)alloc((size_t)HID_ * HID_ * 2);  // Wq^T
        wt[li][1] = (u16*)alloc((size_t)HID_ * HID_ * 2);  // Wk^T
        wt[li][2] = (u16*)alloc((size_t)HID_ * HID_ * 2);  // Wv^T
        wt[li][3] = (u16*)alloc((size_t)HID_ * HID_ * 2);  // Wo^T
        wt[li][4] = (u16*)alloc((size_t)HID_ * FFN_ * 2);  // W1^T [FFN,HID]
        wt[li][5] = (u16*)alloc((size_t)FFN_ * HID_ * 2);  // W2^T [HID,FFN]
    }

    (void)hipMemcpyAsync(x, src_emb, (size_t)ROWS_ * HID_ * 4,
                         hipMemcpyDeviceToDevice, stream);

    const dim3 blk(256);
    // weight prep (once per launch): transpose + f32->bf16
    for (int li = 0; li < 2; ++li) {
        const int base = 8 + 16 * li;
        const int widx[6] = {0, 2, 4, 6, 8, 10};            // Wq Wk Wv Wo W1 W2
        const int Kd[6]   = {HID_, HID_, HID_, HID_, HID_, FFN_};
        const int Nd[6]   = {HID_, HID_, HID_, HID_, FFN_, HID_};
        for (int wi = 0; wi < 6; ++wi) {
            dim3 g(Nd[wi] / 32, Kd[wi] / 32);
            transpose_f2bf_kernel<<<g, blk, 0, stream>>>(
                (const float*)d_in[base + widx[wi]], wt[li][wi], Kd[wi], Nd[wi]);
        }
    }

    const dim3 gemm512(HID_ / BN, ROWS_ / BM);   // (4, 32)
    const dim3 gemmFFN(FFN_ / BN, ROWS_ / BM);   // (16, 32)
    const dim3 attnGrid((B_ * H_ * N_) / 8);
    const dim3 lnGrid(ROWS_ / 8);
    const int  cvt512 = (ROWS_ * HID_ / 8 + 255) / 256;   // 2M elems
    const int  cvtFFN = (ROWS_ * FFN_ / 8 + 255) / 256;   // 8M elems

    for (int li = 0; li < 2; ++li) {
        const int base = 8 + 16 * li;
        const float* bq  = (const float*)d_in[base + 1];
        const float* bk  = (const float*)d_in[base + 3];
        const float* bv  = (const float*)d_in[base + 5];
        const float* bo  = (const float*)d_in[base + 7];
        const float* b1  = (const float*)d_in[base + 9];
        const float* b2  = (const float*)d_in[base + 11];
        const float* l1s = (const float*)d_in[base + 12];
        const float* l1b = (const float*)d_in[base + 13];
        const float* l2s = (const float*)d_in[base + 14];
        const float* l2b = (const float*)d_in[base + 15];

        f2bf_kernel<<<cvt512, blk, 0, stream>>>(x, xb, ROWS_ * HID_ / 8);
        gemm_bf16_wmma<<<gemm512, blk, 0, stream>>>(xb, wt[li][0], bq, qb, ROWS_, HID_, HID_, 0);
        gemm_bf16_wmma<<<gemm512, blk, 0, stream>>>(xb, wt[li][1], bk, kb, ROWS_, HID_, HID_, 0);
        gemm_bf16_wmma<<<gemm512, blk, 0, stream>>>(xb, wt[li][2], bv, vb, ROWS_, HID_, HID_, 0);

        rel_attn_kernel<<<attnGrid, blk, 0, stream>>>(qb, kb, vb, par, bro,
                                                      rel_q, rel_k, rel_v, attn);

        f2bf_kernel<<<cvt512, blk, 0, stream>>>(attn, xb, ROWS_ * HID_ / 8);
        gemm_bf16_wmma<<<gemm512, blk, 0, stream>>>(xb, wt[li][3], bo, tmp, ROWS_, HID_, HID_, 0);
        resid_ln_kernel<<<lnGrid, blk, 0, stream>>>(x, tmp, l1s, l1b, x, ROWS_);

        f2bf_kernel<<<cvt512, blk, 0, stream>>>(x, xb, ROWS_ * HID_ / 8);
        gemm_bf16_wmma<<<gemmFFN, blk, 0, stream>>>(xb, wt[li][4], b1, h1, ROWS_, HID_, FFN_, 1);
        f2bf_kernel<<<cvtFFN, blk, 0, stream>>>(h1, xb, ROWS_ * FFN_ / 8);
        gemm_bf16_wmma<<<gemm512, blk, 0, stream>>>(xb, wt[li][5], b2, tmp, ROWS_, FFN_, HID_, 0);
        resid_ln_kernel<<<lnGrid, blk, 0, stream>>>(x, tmp, l2s, l2b, x, ROWS_);
    }

    resid_ln_kernel<<<lnGrid, blk, 0, stream>>>(x, nullptr, fln_s, fln_b,
                                                (float*)d_out, ROWS_);
}
